// WindowAttnBlock_11003706213290
// MI455X (gfx1250) — matrix operand order repaired and checked by reading, not compiled, not run on hardware
//
#include <hip/hip_runtime.h>

typedef __attribute__((ext_vector_type(16))) __bf16 v16bf;
typedef __attribute__((ext_vector_type(8)))  float  v8f;
typedef unsigned int u32x4 __attribute__((ext_vector_type(4)));
typedef unsigned int u32x8 __attribute__((ext_vector_type(8)));
typedef float        f32x4 __attribute__((ext_vector_type(4)));

#define DEV __device__ __forceinline__

DEV unsigned f2bf(float f) {
  unsigned u = __float_as_uint(f);
  return (u + 0x7FFFu + ((u >> 16) & 1u)) >> 16;
}
DEV unsigned pack2(float a, float b) { return f2bf(a) | (f2bf(b) << 16); }

DEV v8f vzero() { v8f z; for (int i = 0; i < 8; i++) z[i] = 0.f; return z; }

DEV v8f wmma_bf(v16bf a, v16bf b, v8f c) {
  return __builtin_amdgcn_wmma_f32_16x16x32_bf16(false, a, false, b, (short)0, c, false, false);
}

DEV v16bf combine(u32x4 a, u32x4 b) {
  u32x8 c;
  c[0]=a[0]; c[1]=a[1]; c[2]=a[2]; c[3]=a[3];
  c[4]=b[0]; c[5]=b[1]; c[6]=b[2]; c[7]=b[3];
  return __builtin_bit_cast(v16bf, c);
}

// A fragment (16x32 bf16): lane m = lane&15, k = k0 + (lane>>4)*16 + [0..15] contiguous
// (matches B fragment k order)
DEV v16bf ldsA(const char* base, int strideB, int m0, int k0, int lane) {
  int m  = m0 + (lane & 15);
  int kb = (lane >> 4) << 4;
  const u32x4* p = (const u32x4*)(base + m * strideB + ((k0 + kb) << 1));
  return combine(p[0], p[1]);
}
// B fragment (32x16 bf16): lane n = lane&15, k = k0 + (lane>>4)*16 + [0..15] contiguous
DEV v16bf ldsB(const char* base, int strideB, int n0, int k0, int lane) {
  int n  = n0 + (lane & 15);
  int kb = (lane >> 4) << 4;
  const u32x4* p = (const u32x4*)(base + n * strideB + ((k0 + kb) << 1));
  return combine(p[0], p[1]);
}
DEV v16bf glbB(const unsigned short* W, int Kw, int n0, int k0, int lane) {
  int n  = n0 + (lane & 15);
  int kb = (lane >> 4) << 4;
  const u32x4* p = (const u32x4*)(W + (size_t)n * Kw + k0 + kb);
  return combine(p[0], p[1]);
}

// ---------------- weight fp32 -> bf16 conversion ----------------
__global__ __launch_bounds__(256)
void convert_weights(const float* __restrict__ inw, const float* __restrict__ outw,
                     const float* __restrict__ w1,  const float* __restrict__ w2,
                     unsigned short* __restrict__ ws) {
  int i = blockIdx.x * 256 + threadIdx.x;   // grid = 786432 total
  const float* src; int off;
  if      (i < 196608) { src = inw;  off = 0; }
  else if (i < 262144) { src = outw; off = 196608; }
  else if (i < 524288) { src = w1;   off = 262144; }
  else                 { src = w2;   off = 524288; }
  ws[i] = (unsigned short)f2bf(src[i - off]);
}

// ---------------- fused Swin block: one workgroup per 8x8 window ----------------
__global__ __launch_bounds__(256)
void swin_block_kernel(const float* __restrict__ x,
                       const float* __restrict__ ln1g, const float* __restrict__ ln1b,
                       const float* __restrict__ inb,  const float* __restrict__ outb,
                       const float* __restrict__ ln2g, const float* __restrict__ ln2b,
                       const float* __restrict__ b1,   const float* __restrict__ b2,
                       const unsigned short* __restrict__ inw,
                       const unsigned short* __restrict__ outw,
                       const unsigned short* __restrict__ w1,
                       const unsigned short* __restrict__ w2,
                       float* __restrict__ out) {
  // LDS pool layout (bytes); regions overlaid across phases
  constexpr int XN = 0,     XN_S = 528;   // xn / xn2 bf16 [64][264]
  constexpr int QO = 33792, QK_S = 80;    // q bf16 [64][40]
  constexpr int KO = 38912;               // k bf16 [64][40]
  constexpr int VO = 44032, VT_S = 144;   // vT bf16 [32][72]
  constexpr int SO = 48640, SC_S = 260;   // scores f32 [64][65]
  constexpr int AO = 33792, AT_S = 144;   // att bf16 [64][72] (over q/k)
  constexpr int OO = 48640, OH_S = 80;    // o_h bf16 [64][40] (over scores)
  constexpr int PS = 48640;               // LN2 partials [64][2 halves][2 f32]
  constexpr int HO = 33792, H2_S = 144;   // mlp hidden chunk bf16 [64][72]
  __shared__ __align__(16) char pool[65280];

  const int tid = threadIdx.x;
  const int wave = tid >> 5, lane = tid & 31;
  const int hi = lane >> 4, nl = lane & 15;

  const int blk = blockIdx.x;
  const int b  = blk >> 8;
  const int wi = blk & 255;
  const int wr = wi >> 4, wc = wi & 15;

  auto rowbase = [&](int m) -> size_t {
    int ng = ((wr << 3) + (m >> 3)) * 128 + (wc << 3) + (m & 7);
    return ((size_t)b * 16384 + (size_t)ng) << 8;  // * C(256)
  };

  // ---- LN1 + window gather -> xn (bf16) ----
  {
    int t = tid >> 2, c0 = (tid & 3) << 6;
    const float* xp = x + rowbase(t) + c0;
    float s = 0.f, sq = 0.f;
#pragma unroll
    for (int i = 0; i < 16; i++) {
      f32x4 v = ((const f32x4*)xp)[i];
      s  += v[0] + v[1] + v[2] + v[3];
      sq += v[0]*v[0] + v[1]*v[1] + v[2]*v[2] + v[3]*v[3];
    }
    s  += __shfl_xor(s, 1, 32);  s  += __shfl_xor(s, 2, 32);
    sq += __shfl_xor(sq, 1, 32); sq += __shfl_xor(sq, 2, 32);
    float mean = s * (1.f / 256.f);
    float inv  = rsqrtf(sq * (1.f / 256.f) - mean * mean + 1e-5f);
    unsigned* dst = (unsigned*)(pool + XN + t * XN_S + (c0 << 1));
#pragma unroll
    for (int i = 0; i < 16; i++) {
      f32x4 v = ((const f32x4*)xp)[i];
      f32x4 g = ((const f32x4*)(ln1g + c0))[i];
      f32x4 bb = ((const f32x4*)(ln1b + c0))[i];
      dst[2*i]   = pack2((v[0]-mean)*inv*g[0]+bb[0], (v[1]-mean)*inv*g[1]+bb[1]);
      dst[2*i+1] = pack2((v[2]-mean)*inv*g[2]+bb[2], (v[3]-mean)*inv*g[3]+bb[3]);
    }
  }
  __syncthreads();

  const int mt  = wave >> 1;         // M tile 0..3
  const int ntq = wave & 1;          // N half for 32-wide outputs
  const int ntb = (wave & 1) << 3;   // N tile base for 256-wide outputs

  v8f pacc[8];
#pragma unroll
  for (int j = 0; j < 8; j++) pacc[j] = vzero();

  const float scale = 0.1767766953f;  // 1/sqrt(32)

  for (int h = 0; h < 8; ++h) {
    // ---- QKV for this head ----
    {
      v16bf afr[8];
#pragma unroll
      for (int kk = 0; kk < 8; kk++) afr[kk] = ldsA(pool + XN, XN_S, mt << 4, kk << 5, lane);
#pragma unroll
      for (int mat = 0; mat < 3; ++mat) {
        int nrow0 = mat * 256 + h * 32 + (ntq << 4);
        v8f acc = vzero();
#pragma unroll
        for (int kk = 0; kk < 8; kk++) {
          v16bf bf = glbB(inw, 256, nrow0, kk << 5, lane);
          acc = wmma_bf(afr[kk], bf, acc);
        }
        float bias = inb[nrow0 + nl];
        int nloc = (ntq << 4) + nl;
        if (mat < 2) {
          char* dstb = pool + (mat == 0 ? QO : KO);
#pragma unroll
          for (int r = 0; r < 8; r++) {
            int m = (mt << 4) + r + (hi << 3);
            *(unsigned short*)(dstb + m * QK_S + (nloc << 1)) = (unsigned short)f2bf(acc[r] + bias);
          }
        } else {  // v stored transposed: vT[d][s]
#pragma unroll
          for (int r = 0; r < 8; r++) {
            int m = (mt << 4) + r + (hi << 3);
            *(unsigned short*)(pool + VO + nloc * VT_S + (m << 1)) = (unsigned short)f2bf(acc[r] + bias);
          }
        }
      }
    }
    __syncthreads();
    // ---- scores = q @ k^T ----
    {
      v16bf aq = ldsA(pool + QO, QK_S, mt << 4, 0, lane);
#pragma unroll
      for (int j = 0; j < 2; j++) {
        int nt = (ntq << 1) + j;
        v16bf bk = ldsB(pool + KO, QK_S, nt << 4, 0, lane);
        v8f sc = wmma_bf(aq, bk, vzero());
#pragma unroll
        for (int r = 0; r < 8; r++) {
          int m = (mt << 4) + r + (hi << 3);
          int n = (nt << 4) + nl;
          *(float*)(pool + SO + m * SC_S + (n << 2)) = sc[r];
        }
      }
    }
    __syncthreads();
    // ---- softmax -> att (bf16, over dead q/k region) ----
    {
      int t = tid >> 2, s0 = (tid & 3) << 4;
      const float* row = (const float*)(pool + SO + t * SC_S);
      float vals[16]; float mx = -3.0e38f;
#pragma unroll
      for (int i = 0; i < 16; i++) { vals[i] = row[s0 + i] * scale; mx = fmaxf(mx, vals[i]); }
      mx = fmaxf(mx, __shfl_xor(mx, 1, 32));
      mx = fmaxf(mx, __shfl_xor(mx, 2, 32));
      float sum = 0.f;
#pragma unroll
      for (int i = 0; i < 16; i++) { vals[i] = __expf(vals[i] - mx); sum += vals[i]; }
      sum += __shfl_xor(sum, 1, 32); sum += __shfl_xor(sum, 2, 32);
      float inv = 1.f / sum;
      unsigned* arow = (unsigned*)(pool + AO + t * AT_S + (s0 << 1));
#pragma unroll
      for (int i = 0; i < 8; i++) arow[i] = pack2(vals[2*i] * inv, vals[2*i+1] * inv);
    }
    __syncthreads();
    // ---- o_h = att @ v (over dead scores region) ----
    {
      v8f acc = vzero();
#pragma unroll
      for (int kk = 0; kk < 2; kk++) {
        v16bf a  = ldsA(pool + AO, AT_S, mt << 4, kk << 5, lane);
        v16bf bv = ldsB(pool + VO, VT_S, ntq << 4, kk << 5, lane);
        acc = wmma_bf(a, bv, acc);
      }
      int nloc = (ntq << 4) + nl;
#pragma unroll
      for (int r = 0; r < 8; r++) {
        int m = (mt << 4) + r + (hi << 3);
        *(unsigned short*)(pool + OO + m * OH_S + (nloc << 1)) = (unsigned short)f2bf(acc[r]);
      }
    }
    __syncthreads();
    // ---- projection partial: pacc += o_h @ out_w[:, h*32:h*32+32]^T ----
    {
      v16bf a = ldsA(pool + OO, OH_S, mt << 4, 0, lane);
#pragma unroll
      for (int j = 0; j < 8; j++) {
        v16bf bf = glbB(outw, 256, (ntb + j) << 4, h << 5, lane);
        pacc[j] = wmma_bf(a, bf, pacc[j]);
      }
    }
    __syncthreads();
  }

  // ---- x1 = proj + out_b + shortcut -> d_out; LN2 row partial sums ----
  {
    float rs[8], rq[8];
#pragma unroll
    for (int r = 0; r < 8; r++) { rs[r] = 0.f; rq[r] = 0.f; }
#pragma unroll
    for (int j = 0; j < 8; j++) {
      int n = ((ntb + j) << 4) + nl;
      float ob = outb[n];
#pragma unroll
      for (int r = 0; r < 8; r++) {
        int m = (mt << 4) + r + (hi << 3);
        float val = pacc[j][r] + ob + x[rowbase(m) + n];
        out[rowbase(m) + n] = val;
        rs[r] += val; rq[r] += val * val;
      }
    }
#pragma unroll
    for (int r = 0; r < 8; r++) {
#pragma unroll
      for (int o = 1; o < 16; o <<= 1) {
        rs[r] += __shfl_xor(rs[r], o, 32);
        rq[r] += __shfl_xor(rq[r], o, 32);
      }
    }
    if (nl == 0) {
#pragma unroll
      for (int r = 0; r < 8; r++) {
        int m = (mt << 4) + r + (hi << 3);
        float* ps = (float*)(pool + PS + m * 16 + ((wave & 1) << 3));
        ps[0] = rs[r]; ps[1] = rq[r];
      }
    }
  }
  __syncthreads();
  // ---- LN2 -> xn2 (reuse XN region; x1 re-read by owning thread) ----
  {
#pragma unroll
    for (int j = 0; j < 8; j++) {
      int n = ((ntb + j) << 4) + nl;
      float g = ln2g[n], be = ln2b[n];
#pragma unroll
      for (int r = 0; r < 8; r++) {
        int m = (mt << 4) + r + (hi << 3);
        const float* ps = (const float*)(pool + PS + m * 16);
        float s = ps[0] + ps[2], sq = ps[1] + ps[3];
        float mean = s * (1.f / 256.f);
        float inv = rsqrtf(sq * (1.f / 256.f) - mean * mean + 1e-5f);
        float x1v = out[rowbase(m) + n];
        *(unsigned short*)(pool + XN + m * XN_S + (n << 1)) =
            (unsigned short)f2bf((x1v - mean) * inv * g + be);
      }
    }
  }
  __syncthreads();

  // ---- MLP: 16 hidden chunks of 64; GEMM2 accumulates in registers ----
  v8f macc[8];
#pragma unroll
  for (int j = 0; j < 8; j++) macc[j] = vzero();

  for (int c = 0; c < 16; ++c) {
    // GEMM1: hid = gelu(xn2 @ w1_chunk^T + b1)
    {
#pragma unroll
      for (int j2 = 0; j2 < 2; j2++) {
        int nt = (ntq << 1) + j2;
        v8f acc = vzero();
#pragma unroll
        for (int kk = 0; kk < 8; kk++) {
          v16bf a  = ldsA(pool + XN, XN_S, mt << 4, kk << 5, lane);
          v16bf bf = glbB(w1, 256, (c << 6) + (nt << 4), kk << 5, lane);
          acc = wmma_bf(a, bf, acc);
        }
        int nloc = (nt << 4) + nl;
        float bias = b1[(c << 6) + nloc];
#pragma unroll
        for (int r = 0; r < 8; r++) {
          int m = (mt << 4) + r + (hi << 3);
          float v2 = acc[r] + bias;
          v2 = 0.5f * v2 * (1.f + erff(v2 * 0.70710678f));  // exact gelu
          *(unsigned short*)(pool + HO + m * H2_S + (nloc << 1)) = (unsigned short)f2bf(v2);
        }
      }
    }
    __syncthreads();
    // GEMM2 partial: macc += hid @ w2_chunk^T
    {
      v16bf a0 = ldsA(pool + HO, H2_S, mt << 4, 0, lane);
      v16bf a1 = ldsA(pool + HO, H2_S, mt << 4, 32, lane);
#pragma unroll
      for (int j = 0; j < 8; j++) {
        v16bf b0 = glbB(w2, 1024, (ntb + j) << 4, (c << 6), lane);
        v16bf bb = glbB(w2, 1024, (ntb + j) << 4, (c << 6) + 32, lane);
        macc[j] = wmma_bf(a0, b0, macc[j]);
        macc[j] = wmma_bf(a1, bb, macc[j]);
      }
    }
    __syncthreads();
  }

  // ---- final: out = x1 + mlp + b2 (same-thread RMW on d_out) ----
  {
#pragma unroll
    for (int j = 0; j < 8; j++) {
      int n = ((ntb + j) << 4) + nl;
      float bb = b2[n];
#pragma unroll
      for (int r = 0; r < 8; r++) {
        int m = (mt << 4) + r + (hi << 3);
        size_t gi = rowbase(m) + n;
        out[gi] = out[gi] + macc[j][r] + bb;
      }
    }
  }
}

extern "C" void kernel_launch(void* const* d_in, const int* in_sizes, int n_in,
                              void* d_out, int out_size, void* d_ws, size_t ws_size,
                              hipStream_t stream) {
  (void)in_sizes; (void)n_in; (void)out_size; (void)ws_size;
  const float* x    = (const float*)d_in[0];
  const float* ln1g = (const float*)d_in[1];
  const float* ln1b = (const float*)d_in[2];
  const float* inw  = (const float*)d_in[3];
  const float* inb  = (const float*)d_in[4];
  const float* outw = (const float*)d_in[5];
  const float* outb = (const float*)d_in[6];
  const float* ln2g = (const float*)d_in[7];
  const float* ln2b = (const float*)d_in[8];
  const float* w1   = (const float*)d_in[9];
  const float* b1   = (const float*)d_in[10];
  const float* w2   = (const float*)d_in[11];
  const float* b2   = (const float*)d_in[12];
  unsigned short* ws = (unsigned short*)d_ws;  // 786432 bf16 = 1.5 MB

  convert_weights<<<3072, 256, 0, stream>>>(inw, outw, w1, w2, ws);
  swin_block_kernel<<<2048, 256, 0, stream>>>(
      x, ln1g, ln1b, inb, outb, ln2g, ln2b, b1, b2,
      ws, ws + 196608, ws + 262144, ws + 524288, (float*)d_out);
}
